// ZeroGradientSSM4B_17197049053898
// MI455X (gfx1250) — matrix operand order repaired and checked by reading, not compiled, not run on hardware
//
#include <hip/hip_runtime.h>
#include <hip/hip_bf16.h>
#include <cstdint>

// ---------------- model dims ----------------
constexpr int Vv = 32000;
constexpr int Dd = 768;
constexpr int Nn = 16;
constexpr int Ee = 4;
constexpr int Ll = 4;
constexpr int Bb = 2;
constexpr int Ss = 2048;
constexpr int Ff = 4 * Dd;          // 3072
constexpr int Mm = Bb * Ss;         // 4096 tokens
constexpr float EPSf = 1e-5f;

typedef __attribute__((ext_vector_type(16))) _Float16 v16h;
typedef __attribute__((ext_vector_type(8)))  _Float16 v8h;
typedef __attribute__((ext_vector_type(8)))  float    v8f;

enum { EPI_BIAS = 0, EPI_SOFTPLUS = 1, EPI_NOBIAS = 2 };

// ---------------- elementwise helpers ----------------

__global__ void k_embed_gather(const int* __restrict__ x, const float* __restrict__ embed,
                               float* __restrict__ h, _Float16* __restrict__ h16)
{
    int i = blockIdx.x * blockDim.x + threadIdx.x;
    if (i >= Mm * Dd) return;
    int t = i / Dd, d = i - t * Dd;
    float v = embed[(size_t)x[t] * Dd + d];
    h[i] = v;
    h16[i] = (_Float16)v;
}

__global__ void k_f32_to_f16(const float* __restrict__ in, _Float16* __restrict__ out, int n)
{
    int i = blockIdx.x * blockDim.x + threadIdx.x;
    if (i < n) out[i] = (_Float16)in[i];
}

// in: [K,N] f32 row-major  ->  out: [N,K] f16 row-major (K-contiguous "BT")
__global__ void k_conv_tr(const float* __restrict__ in, _Float16* __restrict__ out, int K, int N)
{
    int i = blockIdx.x * blockDim.x + threadIdx.x;
    if (i >= K * N) return;
    int k = i / N, n = i - k * N;
    out[(size_t)n * K + k] = (_Float16)in[i];
}

// ---------------- WMMA fragment loaders (NT layout, K % 32 == 0) ----------------
// A 16x32 fragment: lane half 0 holds K {0..7,16..23}, half 1 holds K {8..15,24..31}
__device__ __forceinline__ v16h load_a_frag(const _Float16* __restrict__ row, int k0, int hf)
{
    v8h lo = *(const v8h*)(row + k0 + hf * 8);
    v8h hi = *(const v8h*)(row + k0 + 16 + hf * 8);
    v16h v;
#pragma unroll
    for (int i = 0; i < 8; ++i) { v[i] = lo[i]; v[i + 8] = hi[i]; }
    return v;
}
// B 32x16 fragment: lane (&15) = column; lane half 0 holds K {0..7,16..23}, half 1 holds K {8..15,24..31}
__device__ __forceinline__ v16h load_b_frag(const _Float16* __restrict__ row, int k0, int hf)
{
    v8h lo = *(const v8h*)(row + k0 + hf * 8);
    v8h hi = *(const v8h*)(row + k0 + 16 + hf * 8);
    v16h v;
#pragma unroll
    for (int i = 0; i < 8; ++i) { v[i] = lo[i]; v[i + 8] = hi[i]; }
    return v;
}

__device__ __forceinline__ v8f wmma(v16h a, v16h b, v8f c)
{
    return __builtin_amdgcn_wmma_f32_16x16x32_f16(false, a, false, b, (short)0, c, false, false);
}

// ---------------- 16x16-per-wave core (used only for N==16 projections) ----------------
__device__ __forceinline__ v8f gemm_tile16(const _Float16* __restrict__ A,
                                           const _Float16* __restrict__ BT,
                                           int K, int m0, int n0)
{
    int lane = threadIdx.x & 31;
    int hf = lane >> 4, r = lane & 15;
    const _Float16* arow = A  + (size_t)(m0 + r) * K;
    const _Float16* brow = BT + (size_t)(n0 + r) * K;
    v8f acc = {};
    for (int k0 = 0; k0 < K; k0 += 32)
        acc = wmma(load_a_frag(arow, k0, hf), load_b_frag(brow, k0, hf), acc);
    return acc;
}

__global__ void k_gemm16_f32out(const _Float16* __restrict__ A, const _Float16* __restrict__ BT,
                                const float* __restrict__ bias, float* __restrict__ C,
                                int Mq, int Nq, int Kq, int mode)
{
    int ntN = Nq >> 4;
    int ntiles = (Mq >> 4) * ntN;
    int waveId = blockIdx.x * (blockDim.x >> 5) + (threadIdx.x >> 5);
    if (waveId >= ntiles) return;                       // wave-uniform
    int tm = waveId / ntN, tn = waveId - tm * ntN;
    v8f acc = gemm_tile16(A, BT, Kq, tm << 4, tn << 4);
    int lane = threadIdx.x & 31, hf = lane >> 4, r = lane & 15;
    int n = (tn << 4) + r;
#pragma unroll
    for (int i = 0; i < 8; ++i) {
        int m = (tm << 4) + hf * 8 + i;
        float v = acc[i];
        if (mode != EPI_NOBIAS) v += bias[n];
        if (mode == EPI_SOFTPLUS) v = (v > 20.f) ? v : log1pf(__expf(v));
        C[(size_t)m * Nq + n] = v;
    }
}

// ---------------- 32x32-per-wave register-blocked core (M,N % 32 == 0) ----------------
// One wave -> 2x2 grid of 16x16 WMMA tiles: 8 b128 loads feed 4 WMMAs per k-step.
struct Acc32 { v8f a00, a01, a10, a11; };

__device__ __forceinline__ Acc32 gemm_tile32(const _Float16* __restrict__ A,
                                             const _Float16* __restrict__ BT,
                                             int K, int m0, int n0)
{
    int lane = threadIdx.x & 31;
    int hf = lane >> 4, r = lane & 15;
    const _Float16* ar0 = A  + (size_t)(m0 + r) * K;
    const _Float16* ar1 = A  + (size_t)(m0 + 16 + r) * K;
    const _Float16* br0 = BT + (size_t)(n0 + r) * K;
    const _Float16* br1 = BT + (size_t)(n0 + 16 + r) * K;
    Acc32 acc;
    acc.a00 = v8f{}; acc.a01 = v8f{}; acc.a10 = v8f{}; acc.a11 = v8f{};
    for (int k0 = 0; k0 < K; k0 += 32) {
        v16h a0 = load_a_frag(ar0, k0, hf);
        v16h a1 = load_a_frag(ar1, k0, hf);
        v16h b0 = load_b_frag(br0, k0, hf);
        v16h b1 = load_b_frag(br1, k0, hf);
        acc.a00 = wmma(a0, b0, acc.a00);
        acc.a01 = wmma(a0, b1, acc.a01);
        acc.a10 = wmma(a1, b0, acc.a10);
        acc.a11 = wmma(a1, b1, acc.a11);
    }
    return acc;
}

// map wave id -> 32x32 block; waves consecutive in N (share A rows via cache)
__device__ __forceinline__ bool tile32_coords(int Mq, int Nq, int& m0, int& n0)
{
    int ntN = Nq >> 5;
    int ntiles = (Mq >> 5) * ntN;
    int waveId = blockIdx.x * (blockDim.x >> 5) + (threadIdx.x >> 5);
    if (waveId >= ntiles) return false;
    int tm = waveId / ntN, tn = waveId - tm * ntN;
    m0 = tm << 5; n0 = tn << 5;
    return true;
}

__device__ __forceinline__ float epi_val(v8f acc, int i) { return acc[i]; }

__global__ void k_gemm32_f32out(const _Float16* __restrict__ A, const _Float16* __restrict__ BT,
                                const float* __restrict__ bias, float* __restrict__ C,
                                int Mq, int Nq, int Kq, int mode)
{
    int m0, n0;
    if (!tile32_coords(Mq, Nq, m0, n0)) return;          // wave-uniform
    Acc32 acc = gemm_tile32(A, BT, Kq, m0, n0);
    int lane = threadIdx.x & 31, hf = lane >> 4, r = lane & 15;
    const v8f* tiles[4] = { &acc.a00, &acc.a01, &acc.a10, &acc.a11 };
#pragma unroll
    for (int t = 0; t < 4; ++t) {
        int ti = t >> 1, tj = t & 1;
        int n = n0 + tj * 16 + r;
        float bv = (mode != EPI_NOBIAS) ? bias[n] : 0.f;
#pragma unroll
        for (int i = 0; i < 8; ++i) {
            int m = m0 + ti * 16 + hf * 8 + i;
            float v = (*tiles[t])[i] + bv;
            if (mode == EPI_SOFTPLUS) v = (v > 20.f) ? v : log1pf(__expf(v));
            C[(size_t)m * Nq + n] = v;
        }
    }
}

__global__ void k_gemm32_silu_f16out(const _Float16* __restrict__ A, const _Float16* __restrict__ BT,
                                     const float* __restrict__ bias, _Float16* __restrict__ C,
                                     int Mq, int Nq, int Kq)
{
    int m0, n0;
    if (!tile32_coords(Mq, Nq, m0, n0)) return;
    Acc32 acc = gemm_tile32(A, BT, Kq, m0, n0);
    int lane = threadIdx.x & 31, hf = lane >> 4, r = lane & 15;
    const v8f* tiles[4] = { &acc.a00, &acc.a01, &acc.a10, &acc.a11 };
#pragma unroll
    for (int t = 0; t < 4; ++t) {
        int ti = t >> 1, tj = t & 1;
        int n = n0 + tj * 16 + r;
        float bv = bias[n];
#pragma unroll
        for (int i = 0; i < 8; ++i) {
            int m = m0 + ti * 16 + hf * 8 + i;
            float v = (*tiles[t])[i] + bv;
            v = v / (1.f + __expf(-v));                  // silu
            C[(size_t)m * Nq + n] = (_Float16)v;
        }
    }
}

// down-projection: write w[m]*(acc + bias) only for rows routed to expert e.
__global__ void k_gemm32_moe(const _Float16* __restrict__ A, const _Float16* __restrict__ BT,
                             const float* __restrict__ bias, float* __restrict__ moe,
                             const int* __restrict__ eidx, const float* __restrict__ ew,
                             int Mq, int Nq, int Kq, int e)
{
    int m0, n0;
    if (!tile32_coords(Mq, Nq, m0, n0)) return;
    Acc32 acc = gemm_tile32(A, BT, Kq, m0, n0);
    int lane = threadIdx.x & 31, hf = lane >> 4, r = lane & 15;
    const v8f* tiles[4] = { &acc.a00, &acc.a01, &acc.a10, &acc.a11 };
#pragma unroll
    for (int t = 0; t < 4; ++t) {
        int ti = t >> 1, tj = t & 1;
        int n = n0 + tj * 16 + r;
        float bv = bias[n];
#pragma unroll
        for (int i = 0; i < 8; ++i) {
            int m = m0 + ti * 16 + hf * 8 + i;
            if (eidx[m] == e)
                moe[(size_t)m * Nq + n] = ((*tiles[t])[i] + bv) * ew[m];
        }
    }
}

// ---------------- sequential selective-scan: one thread per (b, d) ----------------
__global__ void k_scan(const float* __restrict__ h, const float* __restrict__ delta,
                       const float* __restrict__ Bm, const float* __restrict__ Cm,
                       const float* __restrict__ A_log_l, const float* __restrict__ Dp_l,
                       float* __restrict__ ssm)
{
    int i = blockIdx.x * blockDim.x + threadIdx.x;
    if (i >= Bb * Dd) return;
    int b = i / Dd, d = i - b * Dd;
    float Arow[Nn], s[Nn];
#pragma unroll
    for (int n = 0; n < Nn; ++n) { Arow[n] = -__expf(A_log_l[d * Nn + n]); s[n] = 0.f; }
    float dcoef = Dp_l[d];
    for (int t = 0; t < Ss; ++t) {
        size_t row = (size_t)(b * Ss + t);
        float x  = h[row * Dd + d];
        float dl = delta[row * Dd + d];
        const float4* Bt4 = (const float4*)(Bm + row * Nn);
        const float4* Ct4 = (const float4*)(Cm + row * Nn);
        float Bt[Nn], Ct[Nn];
#pragma unroll
        for (int q = 0; q < Nn / 4; ++q) {
            float4 bq = Bt4[q], cq = Ct4[q];
            Bt[q * 4 + 0] = bq.x; Bt[q * 4 + 1] = bq.y; Bt[q * 4 + 2] = bq.z; Bt[q * 4 + 3] = bq.w;
            Ct[q * 4 + 0] = cq.x; Ct[q * 4 + 1] = cq.y; Ct[q * 4 + 2] = cq.z; Ct[q * 4 + 3] = cq.w;
        }
        float y = 0.f;
#pragma unroll
        for (int n = 0; n < Nn; ++n) {
            float barA = __expf(fminf(dl * Arow[n], 2.0f));
            float barB = fminf(fmaxf(dl * Bt[n], -2.0f), 2.0f);
            float sn = barA * s[n] + barB * x;
            sn = fminf(fmaxf(sn, -100.0f), 100.0f);
            s[n] = sn;
            y += sn * Ct[n];
        }
        ssm[row * Dd + d] = y + x * dcoef;
    }
}

// ---------------- router: one thread per token (E = 4) ----------------
__global__ void k_router(const float* __restrict__ flat, const float* __restrict__ Wr_l,
                         const float* __restrict__ br_l, float* __restrict__ w,
                         int* __restrict__ eidx)
{
    int m = blockIdx.x * blockDim.x + threadIdx.x;
    if (m >= Mm) return;
    float acc0 = br_l[0], acc1 = br_l[1], acc2 = br_l[2], acc3 = br_l[3];
    const float4* wr4 = (const float4*)Wr_l;            // [D] rows of float4
    const float* xr = flat + (size_t)m * Dd;
    for (int d = 0; d < Dd; ++d) {
        float xv = xr[d];
        float4 rw = wr4[d];
        acc0 += xv * rw.x; acc1 += xv * rw.y; acc2 += xv * rw.z; acc3 += xv * rw.w;
    }
    float mx = fmaxf(fmaxf(acc0, acc1), fmaxf(acc2, acc3));
    float e0 = __expf(acc0 - mx), e1 = __expf(acc1 - mx);
    float e2 = __expf(acc2 - mx), e3 = __expf(acc3 - mx);
    float denom = e0 + e1 + e2 + e3;
    int best = 0; float bp = e0;
    if (e1 > bp) { bp = e1; best = 1; }
    if (e2 > bp) { bp = e2; best = 2; }
    if (e3 > bp) { bp = e3; best = 3; }
    w[m] = bp / denom;
    eidx[m] = best;
}

// ---------------- residual add + layernorm: one wave per token ----------------
__global__ void k_add_ln(const float* __restrict__ ssm, const float* __restrict__ moe,
                         const float* __restrict__ g, const float* __restrict__ bta,
                         float* __restrict__ hout, _Float16* __restrict__ h16)
{
    int tok  = (blockIdx.x * blockDim.x + threadIdx.x) >> 5;
    int lane = threadIdx.x & 31;
    if (tok >= Mm) return;
    const float* a  = ssm + (size_t)tok * Dd;
    const float* b2 = moe + (size_t)tok * Dd;
    constexpr int PER = Dd / 32;  // 24
    float vals[PER];
    float sum = 0.f, sumsq = 0.f;
#pragma unroll
    for (int j = 0; j < PER; ++j) {
        int d = lane + j * 32;
        float v = a[d] + b2[d];
        vals[j] = v; sum += v; sumsq += v * v;
    }
#pragma unroll
    for (int off = 16; off > 0; off >>= 1) {
        sum   += __shfl_xor(sum, off, 32);
        sumsq += __shfl_xor(sumsq, off, 32);
    }
    float mu = sum / (float)Dd;
    float var = sumsq / (float)Dd - mu * mu;
    float rstd = rsqrtf(var + EPSf);
#pragma unroll
    for (int j = 0; j < PER; ++j) {
        int d = lane + j * 32;
        float v = (vals[j] - mu) * rstd * g[d] + bta[d];
        hout[(size_t)tok * Dd + d] = v;
        h16[(size_t)tok * Dd + d] = (_Float16)v;
    }
}

// ---------------- host ----------------
static inline int cdiv(int a, int b) { return (a + b - 1) / b; }

extern "C" void kernel_launch(void* const* d_in, const int* in_sizes, int n_in,
                              void* d_out, int out_size, void* d_ws, size_t ws_size,
                              hipStream_t stream)
{
    const int*   x     = (const int*)  d_in[0];
    const float* embed = (const float*)d_in[1];
    const float* A_log = (const float*)d_in[2];
    const float* Dp    = (const float*)d_in[3];
    const float* Wd    = (const float*)d_in[4];
    const float* bd    = (const float*)d_in[5];
    const float* WB    = (const float*)d_in[6];
    const float* bB    = (const float*)d_in[7];
    const float* WC    = (const float*)d_in[8];
    const float* bC    = (const float*)d_in[9];
    const float* Wr    = (const float*)d_in[10];
    const float* br    = (const float*)d_in[11];
    const float* Wu    = (const float*)d_in[12];
    const float* bu    = (const float*)d_in[13];
    const float* Wo    = (const float*)d_in[14];
    const float* bo    = (const float*)d_in[15];
    const float* ln_g  = (const float*)d_in[16];
    const float* ln_b  = (const float*)d_in[17];

    char* ws = (char*)d_ws;
    size_t off = 0;
    auto carve = [&](size_t bytes) -> char* {
        char* p = ws + off;
        off += (bytes + 255) & ~(size_t)255;
        return p;
    };
    float*    h      = (float*)   carve((size_t)Mm * Dd * 4);
    _Float16* h16    = (_Float16*)carve((size_t)Mm * Dd * 2);
    float*    delta  = (float*)   carve((size_t)Mm * Dd * 4);
    float*    Bm     = (float*)   carve((size_t)Mm * Nn * 4);
    float*    Cm     = (float*)   carve((size_t)Mm * Nn * 4);
    float*    ssm    = (float*)   carve((size_t)Mm * Dd * 4);
    _Float16* ssm16  = (_Float16*)carve((size_t)Mm * Dd * 2);
    float*    moe    = (float*)   carve((size_t)Mm * Dd * 4);
    float*    wbuf   = (float*)   carve((size_t)Mm * 4);
    int*      idxbuf = (int*)     carve((size_t)Mm * 4);
    _Float16* up16   = (_Float16*)carve((size_t)Mm * Ff * 2);
    _Float16* emb16  = (_Float16*)carve((size_t)Vv * Dd * 2);
    _Float16* WdT    = (_Float16*)carve((size_t)Dd * Dd * 2);
    _Float16* WBT    = (_Float16*)carve((size_t)Nn * Dd * 2);
    _Float16* WCT    = (_Float16*)carve((size_t)Nn * Dd * 2);
    _Float16* WuT    = (_Float16*)carve((size_t)Ff * Dd * 2);
    _Float16* WoT    = (_Float16*)carve((size_t)Dd * Ff * 2);

    const int TB = 256;

    // embedding gather (f32 + f16 copy) and embed -> f16 (for tied logits)
    k_embed_gather<<<cdiv(Mm * Dd, TB), TB, 0, stream>>>(x, embed, h, h16);
    k_f32_to_f16<<<cdiv(Vv * Dd, TB), TB, 0, stream>>>(embed, emb16, Vv * Dd);

    for (int l = 0; l < Ll; ++l) {
        const float* Wd_l = Wd + (size_t)l * Dd * Dd;
        const float* WB_l = WB + (size_t)l * Dd * Nn;
        const float* WC_l = WC + (size_t)l * Dd * Nn;

        // transpose+convert this layer's projection weights
        k_conv_tr<<<cdiv(Dd * Dd, TB), TB, 0, stream>>>(Wd_l, WdT, Dd, Dd);
        k_conv_tr<<<cdiv(Dd * Nn, TB), TB, 0, stream>>>(WB_l, WBT, Dd, Nn);
        k_conv_tr<<<cdiv(Dd * Nn, TB), TB, 0, stream>>>(WC_l, WCT, Dd, Nn);

        // delta = softplus(h @ Wd + bd) ; Bm/Cm = h @ W{B,C} + b{B,C}
        {
            int nt = (Mm / 32) * (Dd / 32);
            k_gemm32_f32out<<<cdiv(nt, 8), TB, 0, stream>>>(h16, WdT, bd + l * Dd, delta,
                                                            Mm, Dd, Dd, EPI_SOFTPLUS);
            int nt2 = (Mm / 16) * (Nn / 16);
            k_gemm16_f32out<<<cdiv(nt2, 8), TB, 0, stream>>>(h16, WBT, bB + l * Nn, Bm,
                                                             Mm, Nn, Dd, EPI_BIAS);
            k_gemm16_f32out<<<cdiv(nt2, 8), TB, 0, stream>>>(h16, WCT, bC + l * Nn, Cm,
                                                             Mm, Nn, Dd, EPI_BIAS);
        }

        // sequential SSM scan
        k_scan<<<cdiv(Bb * Dd, TB), TB, 0, stream>>>(h, delta, Bm, Cm,
                                                     A_log + (size_t)l * Dd * Nn,
                                                     Dp + (size_t)l * Dd, ssm);

        // router on ssm output
        k_router<<<cdiv(Mm, TB), TB, 0, stream>>>(ssm, Wr + (size_t)l * Dd * Ee,
                                                  br + (size_t)l * Ee, wbuf, idxbuf);
        k_f32_to_f16<<<cdiv(Mm * Dd, TB), TB, 0, stream>>>(ssm, ssm16, Mm * Dd);

        // dense MoE (reference computes all experts, selects top-1)
        for (int e = 0; e < Ee; ++e) {
            const float* Wu_le = Wu + ((size_t)l * Ee + e) * Dd * Ff;
            const float* Wo_le = Wo + ((size_t)l * Ee + e) * Ff * Dd;
            const float* bu_le = bu + ((size_t)l * Ee + e) * Ff;
            const float* bo_le = bo + ((size_t)l * Ee + e) * Dd;
            k_conv_tr<<<cdiv(Dd * Ff, TB), TB, 0, stream>>>(Wu_le, WuT, Dd, Ff);
            k_conv_tr<<<cdiv(Ff * Dd, TB), TB, 0, stream>>>(Wo_le, WoT, Ff, Dd);

            int ntu = (Mm / 32) * (Ff / 32);
            k_gemm32_silu_f16out<<<cdiv(ntu, 8), TB, 0, stream>>>(ssm16, WuT, bu_le, up16,
                                                                  Mm, Ff, Dd);
            int ntd = (Mm / 32) * (Dd / 32);
            k_gemm32_moe<<<cdiv(ntd, 8), TB, 0, stream>>>(up16, WoT, bo_le, moe,
                                                          idxbuf, wbuf, Mm, Dd, Ff, e);
        }

        // out = ssm + moe ; layernorm -> h (f32) + h16 (f16)
        k_add_ln<<<cdiv(Mm * 32, TB), TB, 0, stream>>>(ssm, moe, ln_g + l * Dd,
                                                       ln_b + l * Dd, h, h16);
    }

    // tied-embedding logits: h @ embed.T  (embed is [V,D] row-major == ready-made BT)
    {
        int nt = (Mm / 32) * (Vv / 32);
        k_gemm32_f32out<<<cdiv(nt, 8), TB, 0, stream>>>(h16, emb16, nullptr, (float*)d_out,
                                                        Mm, Vv, Dd, EPI_NOBIAS);
    }
    (void)in_sizes; (void)n_in; (void)out_size; (void)ws_size;
}
